// EGNNLayer_83854941487131
// MI455X (gfx1250) — matrix operand order repaired and checked by reading, not compiled, not run on hardware
//
#include <hip/hip_runtime.h>

#define DIMF 128

typedef __attribute__((ext_vector_type(16))) __bf16 v16bf;
typedef __attribute__((ext_vector_type(8)))  __bf16 v8bf;
typedef __attribute__((ext_vector_type(8)))  float  v8f;

static __device__ __forceinline__ v16bf ld16(const __bf16* p0, const __bf16* p1) {
    v8bf lo = *(const v8bf*)p0;
    v8bf hi = *(const v8bf*)p1;
    return __builtin_shufflevector(lo, hi, 0,1,2,3,4,5,6,7,8,9,10,11,12,13,14,15);
}

// fast silu: x * rcp(1+exp(-x))  -> v_exp_f32, v_add_f32, v_rcp_f32, v_mul_f32
static __device__ __forceinline__ float silu(float x) {
    return x * __builtin_amdgcn_rcpf(1.0f + __expf(-x));
}

static __device__ __forceinline__ v8f wmma_bf16(v16bf a, v16bf b, v8f c) {
    return __builtin_amdgcn_wmma_f32_16x16x32_bf16(false, a, false, b, (short)0, c,
                                                   false, false);
}

// ---------------------------------------------------------------------------
// Prep: transpose weights to bf16 [n][k] layout, stash radial row, etc.
// ---------------------------------------------------------------------------
__global__ void prep_weights(const float* __restrict__ e_w1,
                             const float* __restrict__ e_w2,
                             const float* __restrict__ c_w1,
                             const float* __restrict__ n_w1,
                             const float* __restrict__ n_w2,
                             __bf16* __restrict__ w1t,  __bf16* __restrict__ w2t,
                             __bf16* __restrict__ cw1t, __bf16* __restrict__ nw1t,
                             __bf16* __restrict__ nw2t, float* __restrict__ w256) {
    int i = blockIdx.x * blockDim.x + threadIdx.x;
    if (i < 128 * 256) {
        int n = i >> 8, k = i & 255;
        w1t[i]  = (__bf16)e_w1[k * 128 + n];   // e_w1 is [257][128]
        nw1t[i] = (__bf16)n_w1[k * 128 + n];   // n_w1 is [256][128]
    }
    if (i < 128 * 128) {
        int n = i >> 7, k = i & 127;
        w2t[i]  = (__bf16)e_w2[k * 128 + n];
        cw1t[i] = (__bf16)c_w1[k * 128 + n];
        nw2t[i] = (__bf16)n_w2[k * 128 + n];
    }
    if (i < 128) w256[i] = e_w1[256 * 128 + i];  // radial row of e_w1
}

__global__ void prep_h(const float* __restrict__ h, const float* __restrict__ coords,
                       __bf16* __restrict__ hbf, float* __restrict__ agg,
                       float* __restrict__ cout, int nh, int nc) {
    int i = blockIdx.x * blockDim.x + threadIdx.x;
    if (i < nh) { hbf[i] = (__bf16)h[i]; agg[i] = 0.0f; }
    if (i < nc) cout[i] = coords[i];
}

// ---------------------------------------------------------------------------
// Fused edge kernel: 16 edges per wave, 4 waves per block.
//   edge_feat -> silu(x@W1+b1) -> silu(x@W2+b2) = msg
//   silu(msg@Cw1+cb1)@c_w2 = coord weight
//   atomics: agg[row] += msg ; coords_out[row] += cd*w
// ---------------------------------------------------------------------------
struct __align__(16) EdgeSlot {
    int    rows[16];
    int    cols[16];
    float  rad[16];
    float  cd[16][3];
    __bf16 act1[16 * DIMF];
    __bf16 act2[16 * DIMF];
    __bf16 act3[16 * DIMF];
};

__global__ void __launch_bounds__(128) egnn_edge_kernel(
        const float* __restrict__ coords, const int* __restrict__ eidx,
        const float* __restrict__ e_b1,   const float* __restrict__ e_b2,
        const float* __restrict__ c_b1,   const float* __restrict__ c_w2,
        const __bf16* __restrict__ w1t,   // [128][256]
        const __bf16* __restrict__ w2t,   // [128][128]
        const __bf16* __restrict__ cw1t,  // [128][128]
        const float* __restrict__ w256,   // [128]
        const __bf16* __restrict__ hbf,   // [N][128]
        float* __restrict__ agg,          // [N][128]
        float* __restrict__ coords_out,   // [N][3]
        int E) {
    __shared__ EdgeSlot slot[4];
    const int wid  = threadIdx.x >> 5;
    const int lane = threadIdx.x & 31;
    const int hf   = lane >> 4;     // lane half: selects K sub-runs
    const int nl   = lane & 15;     // N (and A-row) index within tile
    EdgeSlot& S = slot[wid];

    const int tile = blockIdx.x * 4 + wid;
    const int e0   = tile * 16;
    if (e0 >= E) return;            // wave-uniform, EXEC stays all-ones for WMMA

    if (lane < 16) {
        int e = e0 + lane;
        int r = eidx[e];
        int c = eidx[E + e];
        S.rows[lane] = r;
        S.cols[lane] = c;
        float dx = coords[r * 3 + 0] - coords[c * 3 + 0];
        float dy = coords[r * 3 + 1] - coords[c * 3 + 1];
        float dz = coords[r * 3 + 2] - coords[c * 3 + 2];
        S.cd[lane][0] = dx; S.cd[lane][1] = dy; S.cd[lane][2] = dz;
        S.rad[lane] = dx * dx + dy * dy + dz * dz;
    }
    // wave-private LDS: in-order DS pipeline, compiler inserts s_wait_dscnt

    const __bf16* rowp = hbf + (size_t)S.rows[nl] * DIMF;  // A-row gather ptrs
    const __bf16* colp = hbf + (size_t)S.cols[nl] * DIMF;

    int   myrow[8];
    float myrad[8];
#pragma unroll
    for (int i = 0; i < 8; ++i) {   // C-layout rows owned by this lane
        myrow[i] = S.rows[i + 8 * hf];
        myrad[i] = S.rad[i + 8 * hf];
    }

    // ---- Layer 1: [16x257] @ [257x128], radial column as rank-1 update ----
    // A-fragments loaded once (global gather), reused across all 8 N-tiles.
    v16bf a1[8];
#pragma unroll
    for (int kt = 0; kt < 8; ++kt) {
        const __bf16* src = (kt < 4) ? rowp : colp;   // h[row] ‖ h[col]
        const int kb = (kt & 3) * 32;
        a1[kt] = ld16(src + kb + 16 * hf, src + kb + 16 * hf + 8);
    }
#pragma unroll 1
    for (int nt = 0; nt < 8; ++nt) {
        const int   n  = nt * 16 + nl;
        const float bv = e_b1[n];
        const float wr = w256[n];
        v8f acc;
#pragma unroll
        for (int i = 0; i < 8; ++i) acc[i] = bv + myrad[i] * wr;
        const __bf16* wbase = w1t + (size_t)n * 256;
#pragma unroll
        for (int kt = 0; kt < 8; ++kt) {
            const __bf16* wp = wbase + kt * 32 + 16 * hf;
            v16bf b = ld16(wp, wp + 8);
            acc = wmma_bf16(a1[kt], b, acc);
        }
#pragma unroll
        for (int i = 0; i < 8; ++i)
            S.act1[(i + 8 * hf) * DIMF + n] = (__bf16)silu(acc[i]);
    }

    // ---- Layer 2: msg = silu(act1 @ e_w2 + b2); scatter into agg ----
    {
        const __bf16* arow = S.act1 + nl * DIMF;
        v16bf a2[4];
#pragma unroll
        for (int kt = 0; kt < 4; ++kt)
            a2[kt] = ld16(arow + kt * 32 + 16 * hf, arow + kt * 32 + 16 * hf + 8);
#pragma unroll 1
        for (int nt = 0; nt < 8; ++nt) {
            const int n = nt * 16 + nl;
            const float bv = e_b2[n];
            v8f acc;
#pragma unroll
            for (int i = 0; i < 8; ++i) acc[i] = bv;
            const __bf16* wbase = w2t + (size_t)n * 128;
#pragma unroll
            for (int kt = 0; kt < 4; ++kt) {
                const __bf16* wp = wbase + kt * 32 + 16 * hf;
                v16bf b = ld16(wp, wp + 8);
                acc = wmma_bf16(a2[kt], b, acc);
            }
#pragma unroll
            for (int i = 0; i < 8; ++i) {
                float x = silu(acc[i]);
                S.act2[(i + 8 * hf) * DIMF + n] = (__bf16)x;
                atomicAdd(&agg[(size_t)myrow[i] * DIMF + n], x);  // segment_sum
            }
        }
    }

    // ---- Layer 3: silu(msg @ c_w1 + c_b1) ----
    {
        const __bf16* arow = S.act2 + nl * DIMF;
        v16bf a3[4];
#pragma unroll
        for (int kt = 0; kt < 4; ++kt)
            a3[kt] = ld16(arow + kt * 32 + 16 * hf, arow + kt * 32 + 16 * hf + 8);
#pragma unroll 1
        for (int nt = 0; nt < 8; ++nt) {
            const int n = nt * 16 + nl;
            const float bv = c_b1[n];
            v8f acc;
#pragma unroll
            for (int i = 0; i < 8; ++i) acc[i] = bv;
            const __bf16* wbase = cw1t + (size_t)n * 128;
#pragma unroll
            for (int kt = 0; kt < 4; ++kt) {
                const __bf16* wp = wbase + kt * 32 + 16 * hf;
                v16bf b = ld16(wp, wp + 8);
                acc = wmma_bf16(a3[kt], b, acc);
            }
#pragma unroll
            for (int i = 0; i < 8; ++i)
                S.act3[(i + 8 * hf) * DIMF + n] = (__bf16)silu(acc[i]);
        }
    }

    // ---- coord weight: 128-dot with c_w2, then scatter into coords_out ----
    if (lane < 16) {
        float w = 0.0f;
        const __bf16* a3 = S.act3 + lane * DIMF;
#pragma unroll 4
        for (int k = 0; k < 128; ++k) w += (float)a3[k] * c_w2[k];
        int r = S.rows[lane];
#pragma unroll
        for (int j = 0; j < 3; ++j)
            atomicAdd(&coords_out[(size_t)r * 3 + j], S.cd[lane][j] * w);
    }
}

// ---------------------------------------------------------------------------
// Node kernel: h_new = silu([h‖agg] @ n_w1 + b1) @ n_w2 + b2 + h
// 16 nodes per wave, 4 waves per block.
// ---------------------------------------------------------------------------
struct __align__(16) NodeSlot { __bf16 act[16 * DIMF]; };

__global__ void __launch_bounds__(128) egnn_node_kernel(
        const float* __restrict__ h,    const float* __restrict__ agg,
        const float* __restrict__ n_b1, const float* __restrict__ n_b2,
        const __bf16* __restrict__ w1t,  // [128][256]
        const __bf16* __restrict__ w2t,  // [128][128]
        const __bf16* __restrict__ hbf,
        float* __restrict__ hout, int Nn) {
    __shared__ NodeSlot slot[4];
    const int wid  = threadIdx.x >> 5;
    const int lane = threadIdx.x & 31;
    const int hf   = lane >> 4;
    const int nl   = lane & 15;
    NodeSlot& S = slot[wid];

    const int t = blockIdx.x * 4 + wid;
    if (t * 16 >= Nn) return;
    const int nodeA = t * 16 + nl;
    const __bf16* hp = hbf + (size_t)nodeA * DIMF;
    const float*  ap = agg + (size_t)nodeA * DIMF;

    // A-fragments for [h ‖ agg], loaded once, reused across 8 N-tiles.
    v16bf a1[8];
#pragma unroll
    for (int kt = 0; kt < 4; ++kt) {
        const int kb = kt * 32;
        a1[kt] = ld16(hp + kb + 16 * hf, hp + kb + 16 * hf + 8);
    }
#pragma unroll
    for (int kt = 4; kt < 8; ++kt) {   // agg half: f32 -> bf16 on the fly
        const int kb = (kt - 4) * 32;
        const float* p0 = ap + kb + 16 * hf;
        const float* p1 = ap + kb + 16 * hf + 8;
        v16bf a;
#pragma unroll
        for (int j = 0; j < 8; ++j) {
            a[j]     = (__bf16)p0[j];
            a[8 + j] = (__bf16)p1[j];
        }
        a1[kt] = a;
    }

#pragma unroll 1
    for (int nt = 0; nt < 8; ++nt) {
        const int n = nt * 16 + nl;
        const float bv = n_b1[n];
        v8f acc;
#pragma unroll
        for (int i = 0; i < 8; ++i) acc[i] = bv;
        const __bf16* wbase = w1t + (size_t)n * 256;
#pragma unroll
        for (int kt = 0; kt < 8; ++kt) {
            const __bf16* wp = wbase + kt * 32 + 16 * hf;
            v16bf b = ld16(wp, wp + 8);
            acc = wmma_bf16(a1[kt], b, acc);
        }
#pragma unroll
        for (int i = 0; i < 8; ++i)
            S.act[(i + 8 * hf) * DIMF + n] = (__bf16)silu(acc[i]);
    }

    {
        const __bf16* arow = S.act + nl * DIMF;
        v16bf a2[4];
#pragma unroll
        for (int kt = 0; kt < 4; ++kt)
            a2[kt] = ld16(arow + kt * 32 + 16 * hf, arow + kt * 32 + 16 * hf + 8);
#pragma unroll 1
        for (int nt = 0; nt < 8; ++nt) {
            const int n = nt * 16 + nl;
            const float bv = n_b2[n];
            v8f acc;
#pragma unroll
            for (int i = 0; i < 8; ++i) acc[i] = bv;
            const __bf16* wbase = w2t + (size_t)n * 128;
#pragma unroll
            for (int kt = 0; kt < 4; ++kt) {
                const __bf16* wp = wbase + kt * 32 + 16 * hf;
                v16bf b = ld16(wp, wp + 8);
                acc = wmma_bf16(a2[kt], b, acc);
            }
#pragma unroll
            for (int i = 0; i < 8; ++i) {     // residual +h fused in epilogue
                const int node = t * 16 + i + 8 * hf;
                float x = acc[i] + h[(size_t)node * DIMF + n];
                hout[(size_t)node * DIMF + n] = x;
            }
        }
    }
}

// ---------------------------------------------------------------------------
extern "C" void kernel_launch(void* const* d_in, const int* in_sizes, int n_in,
                              void* d_out, int out_size, void* d_ws, size_t ws_size,
                              hipStream_t stream) {
    (void)n_in; (void)out_size; (void)ws_size;
    const float* h      = (const float*)d_in[0];
    const float* coords = (const float*)d_in[1];
    const int*   eidx   = (const int*)d_in[2];
    const float* e_w1   = (const float*)d_in[3];
    const float* e_b1   = (const float*)d_in[4];
    const float* e_w2   = (const float*)d_in[5];
    const float* e_b2   = (const float*)d_in[6];
    const float* n_w1   = (const float*)d_in[7];
    const float* n_b1   = (const float*)d_in[8];
    const float* n_w2   = (const float*)d_in[9];
    const float* n_b2   = (const float*)d_in[10];
    const float* c_w1   = (const float*)d_in[11];
    const float* c_b1   = (const float*)d_in[12];
    const float* c_w2   = (const float*)d_in[13];

    const int N = in_sizes[0] / DIMF;
    const int E = in_sizes[2] / 2;

    // workspace layout (bytes)
    char* ws = (char*)d_ws;
    __bf16* w1t  = (__bf16*)(ws + 0);        // 128*256*2 = 65536
    __bf16* w2t  = (__bf16*)(ws + 65536);    // 32768
    __bf16* cw1t = (__bf16*)(ws + 98304);    // 32768
    __bf16* nw1t = (__bf16*)(ws + 131072);   // 65536
    __bf16* nw2t = (__bf16*)(ws + 196608);   // 32768
    float*  w256 = (float*)(ws + 229376);    // 512
    __bf16* hbf  = (__bf16*)(ws + 229888);   // N*128*2
    float*  agg  = (float*)(ws + 229888 + (size_t)N * DIMF * 2);  // N*128*4

    float* hout = (float*)d_out;
    float* cout = hout + (size_t)N * DIMF;

    prep_weights<<<128, 256, 0, stream>>>(e_w1, e_w2, c_w1, n_w1, n_w2,
                                          w1t, w2t, cw1t, nw1t, nw2t, w256);
    prep_h<<<(N * DIMF + 255) / 256, 256, 0, stream>>>(h, coords, hbf, agg, cout,
                                                       N * DIMF, N * 3);

    const int etiles = (E + 15) / 16;
    egnn_edge_kernel<<<(etiles + 3) / 4, 128, 0, stream>>>(
        coords, eidx, e_b1, e_b2, c_b1, c_w2,
        w1t, w2t, cw1t, w256, hbf, agg, cout, E);

    const int ntiles = (N + 15) / 16;
    egnn_node_kernel<<<(ntiles + 3) / 4, 128, 0, stream>>>(
        h, agg, n_b1, n_b2, nw1t, nw2t, hbf, hout, N);
}
